// MyRNN_cat2_84490596647083
// MI455X (gfx1250) — compile-verified, matrix operand order checked by reading, not run on hardware
//
#include <hip/hip_runtime.h>

// Problem constants
#define Bsz   64
#define Tlen  512
#define DXc   256
#define Hc    1024
#define DOUTc 256
#define XINW  768          // DX + 2*DOUT

#define NWG   32           // persistent workgroups (must be co-resident)
#define TPB   256          // 8 wave32 per WG -> 256 waves total
#define NWAVE (NWG * TPB / 32)

// LDS layout (dynamic): padded slabs, +8B pad per 1024B stored (bank de-conflict)
#define A_OFF   0u
#define A_BYTES 132096u    // 64 rows x 1024 bf16 + pads
#define B_OFF   132096u
#define B_BYTES 66048u     // 32 rows x 1024 bf16 + pads (x2 regions, double buffered)
#define LDS_TOTAL (A_BYTES + 2u * B_BYTES)   // 264192 < 320KB/WGP

typedef __attribute__((ext_vector_type(16))) __bf16 v16bf;
typedef __attribute__((ext_vector_type(8)))  __bf16 v8bf;
typedef __attribute__((ext_vector_type(8)))  float  v8f;
typedef unsigned int u32x4 __attribute__((ext_vector_type(4)));
typedef int          i32x4 __attribute__((ext_vector_type(4)));
typedef int          i32x8 __attribute__((ext_vector_type(8)));

#if __has_builtin(__builtin_amdgcn_tensor_load_to_lds)
#define USE_TDM 1
#endif

struct Params {
  const float* x;      // [B,T,DX]
  const float* y0;     // [B,1,DOUT]
  const float* y1;     // [B,1,DOUT]
  const int*   xlen;   // [B]
  const float *b_e0, *b_e1, *b_e2;
  const float *b_ih, *b_hh;
  const float *b_o0, *b_o1, *b_o2;
  const __bf16 *W0t, *W1t, *W2t, *Wih, *Whh, *O0t, *O1t, *O2t; // [N][K] bf16
  __bf16 *xin;                          // [B][768]
  __bf16 *e1, *e2, *e3, *h0, *h1, *hr, *o1, *o2;  // [B][H]
  float* out;                           // [B,T,DOUT]
  unsigned* bar;
};

__device__ __forceinline__ unsigned padoff(unsigned stored) {
  return stored + ((stored >> 10) << 3);   // TDM pad: 2 dwords per 256 dwords
}

// ---- TDM: DMA a [rows x rowElems] bf16 block (contiguous) into padded LDS ----
// D# per CDNA5 ISA ch.8: 2-D tile, data_size=2B, pad_interval=256dw, pad_amount=2dw
__device__ __forceinline__ void tdm_issue(unsigned char* lds, unsigned ldsOff,
                                          const __bf16* gsrc,
                                          unsigned rowElems, unsigned rows) {
#ifdef USE_TDM
  unsigned long long ga = (unsigned long long)(uintptr_t)gsrc;
  unsigned ldsAbs = (unsigned)(uintptr_t)lds + ldsOff;
  u32x4 g0;
  g0[0] = 1u;                                   // count=1, user mode
  g0[1] = ldsAbs;                               // lds_addr
  g0[2] = (unsigned)(ga & 0xFFFFFFFFu);         // global_addr[31:0]
  g0[3] = (unsigned)((ga >> 32) & 0x1FFFFFFu) | (2u << 30);  // addr[56:32]|type=2
  i32x8 g1;
  g1[0] = (int)((1u << 16)      // data_size = 2 bytes
              | (1u << 20)      // pad_enable
              | (7u << 22)      // pad_interval: 256 dwords
              | (1u << 25));    // pad_amount: 2 dwords
  g1[1] = (int)((rowElems & 0xFFFFu) << 16);          // tensor_dim0 lo16
  g1[2] = (int)((rowElems >> 16) | ((rows & 0xFFFFu) << 16)); // dim0 hi | tensor_dim1
  g1[3] = (int)((rowElems & 0xFFFFu) << 16);          // tile_dim0 = rowElems
  g1[4] = (int)rows;                                  // tile_dim1 = rows, tile_dim2=0
  g1[5] = (int)rowElems;                              // tensor_dim0_stride (contig)
  g1[6] = 0;
  g1[7] = 0;
  i32x4 z4 = {0, 0, 0, 0};
#if __clang_major__ >= 23
  i32x8 z8 = {0, 0, 0, 0, 0, 0, 0, 0};
  __builtin_amdgcn_tensor_load_to_lds(g0, g1, z4, z4, z8, 0);
#else
  __builtin_amdgcn_tensor_load_to_lds(g0, g1, z4, z4, 0);
#endif
#endif
}

#ifndef USE_TDM
__device__ __forceinline__ void coop_copy(unsigned char* lds, unsigned off,
                                          const __bf16* src, unsigned n) {
  for (unsigned i = threadIdx.x; i < n; i += TPB) {
    *(__bf16*)(lds + off + padoff(i * 2u)) = src[i];
  }
}
#endif

// Per stage: DMA this stage's A slab and NEXT stage's B slab (into region breg^1),
// then wait tensorcnt<=1: in-order TDM completion => current B + A are resident
// while the next B keeps streaming during compute and the global barrier.
__device__ __forceinline__ void stage_dma(unsigned char* lds, int breg,
                                          const __bf16* Asrc, unsigned aRowElems,
                                          unsigned aRows,
                                          const __bf16* BnextSrc, unsigned bNextRowElems) {
  __syncthreads();   // all waves done reading LDS before overwrite
#ifdef USE_TDM
  if (threadIdx.x < 32) {      // one wave issues TDM ops (EXEC ignored by TDM)
    tdm_issue(lds, A_OFF, Asrc, aRowElems, aRows);
    tdm_issue(lds, B_OFF + ((breg ^ 1) ? B_BYTES : 0u), BnextSrc, bNextRowElems, 32);
    __builtin_amdgcn_s_wait_tensorcnt(1);
  }
#else
  coop_copy(lds, A_OFF, Asrc, aRowElems * aRows);
  coop_copy(lds, B_OFF + ((breg ^ 1) ? B_BYTES : 0u), BnextSrc, bNextRowElems * 32u);
#endif
  __syncthreads();
}

// ---- WMMA fragment from padded LDS slab (16-bit A/B layout) ----
__device__ __forceinline__ v16bf frag_lds(const unsigned char* lds, unsigned region,
                                          int row0, int K, int k, int lane) {
  int r  = row0 + (lane & 15);
  int kb = (lane >> 4) << 3;
  unsigned s0 = (unsigned)((r * K + k + kb) * 2);
  unsigned s1 = s0 + 32u;
  v8bf lo = *(const v8bf*)(lds + region + padoff(s0));
  v8bf hi = *(const v8bf*)(lds + region + padoff(s1));
  return __builtin_shufflevector(lo, hi, 0,1,2,3,4,5,6,7,8,9,10,11,12,13,14,15);
}

// One 16x16 tile, A and B both from LDS; depth-2 software pipeline so the
// scheduler can issue partial DS waits instead of a full drain per WMMA.
__device__ __forceinline__ void gemm_tile(const unsigned char* lds, int aRow0,
                                          unsigned bReg, int bRow0, int K,
                                          int lane, v8f& acc) {
  v16bf a0 = frag_lds(lds, A_OFF, aRow0, K, 0,  lane);
  v16bf b0 = frag_lds(lds, bReg,  bRow0, K, 0,  lane);
  v16bf a1 = frag_lds(lds, A_OFF, aRow0, K, 32, lane);
  v16bf b1 = frag_lds(lds, bReg,  bRow0, K, 32, lane);
  for (int k = 64; k < K; k += 32) {
    v16bf a2 = frag_lds(lds, A_OFF, aRow0, K, k, lane);
    v16bf b2 = frag_lds(lds, bReg,  bRow0, K, k, lane);
    acc = __builtin_amdgcn_wmma_f32_16x16x32_bf16(false, a0, false, b0,
                                                  (short)0, acc, false, false);
    a0 = a1; b0 = b1; a1 = a2; b1 = b2;
  }
  acc = __builtin_amdgcn_wmma_f32_16x16x32_bf16(false, a0, false, b0,
                                                (short)0, acc, false, false);
  acc = __builtin_amdgcn_wmma_f32_16x16x32_bf16(false, a1, false, b1,
                                                (short)0, acc, false, false);
}

// C/D layout: VGPR r: lanes 0-15 -> (M=r, N=lane), lanes 16-31 -> (M=8+r, N=lane-16)
__device__ __forceinline__ void store_act(__bf16* out, int stride, int m0, int n0,
                                          int lane, v8f acc, const float* bias, bool relu) {
  int n  = n0 + (lane & 15);
  float bv = bias[n];
  int rb = m0 + ((lane >> 4) << 3);
#pragma unroll
  for (int r = 0; r < 8; ++r) {
    float v = acc[r] + bv;
    if (relu) v = fmaxf(v, 0.f);
    out[(size_t)(rb + r) * stride + n] = (__bf16)v;
  }
}

// Device-wide barrier: monotonic counter. s_cluster_barrier is a NOP when not
// launched as a cluster but exercises the CDNA5 cluster-sync path.
__device__ __forceinline__ void gbar(unsigned* bar, unsigned target) {
  __threadfence();
  __syncthreads();
  __builtin_amdgcn_s_cluster_barrier();
  if (threadIdx.x == 0) {
    __hip_atomic_fetch_add(bar, 1u, __ATOMIC_RELEASE, __HIP_MEMORY_SCOPE_AGENT);
    while (__hip_atomic_load(bar, __ATOMIC_ACQUIRE, __HIP_MEMORY_SCOPE_AGENT) < target) {
      __builtin_amdgcn_s_sleep(1);
    }
  }
  __syncthreads();
  __threadfence();
}

__global__ __launch_bounds__(TPB, 1) void rnn_persistent(Params p) {
  extern __shared__ unsigned char lds_raw[];   // LDS_TOTAL bytes

  const int lane = threadIdx.x & 31;
  const int w    = threadIdx.x >> 5;           // wave in block: 0..7 (4m x 2n)
  const int tid  = blockIdx.x * TPB + threadIdx.x;
  const int wid  = tid >> 5;
  unsigned bars = 0;

  // One WG owns 2 consecutive n-tiles for ALL 4 m-tiles -> its B rows (32xK)
  // are DMA'd exactly once per device per stage.
  const int nrow  = (int)blockIdx.x * 32;      // first weight row owned by WG
  const int aRow0 = (w & 3) << 4;              // m-tile of this wave
  const int bRow0 = (w >> 2) << 4;             // n-tile (within WG B slab)
  const int n0    = nrow + bRow0;              // global output column base

  // ---- preamble: xin(t=0) = [x_0 | y_1 | y_0]; h(0) = 0 ----
  for (int i = tid; i < Bsz * XINW; i += NWG * TPB) {
    int b = i / XINW, c = i - b * XINW;
    float v;
    if (c < DXc)              v = p.x[((size_t)b * Tlen + 0) * DXc + c];
    else if (c < DXc + DOUTc) v = p.y1[b * DOUTc + (c - DXc)];
    else                      v = p.y0[b * DOUTc + (c - DXc - DOUTc)];
    p.xin[i] = (__bf16)v;
  }
  for (int i = tid; i < Bsz * Hc; i += NWG * TPB) p.h0[i] = (__bf16)0.f;

  // Bootstrap the weight pipeline: B(W0) -> region 0 (overlaps with barrier)
  int breg = 0;
#ifdef USE_TDM
  if (threadIdx.x < 32)
    tdm_issue(lds_raw, B_OFF, p.W0t + (size_t)nrow * XINW, XINW, 32);
#else
  coop_copy(lds_raw, B_OFF, p.W0t + (size_t)nrow * XINW, 32u * XINW);
#endif
  gbar(p.bar, ++bars * NWG);

  for (int t = 0; t < Tlen; ++t) {
    const __bf16* hp = (t & 1) ? p.h1 : p.h0;
    __bf16*       hn = (t & 1) ? p.h0 : p.h1;
    unsigned bcur;

    // S1: e1 = relu(xin @ W0 + b0)     [K=768]
    stage_dma(lds_raw, breg, p.xin, XINW, 64, p.W1t + (size_t)nrow * Hc, Hc);
    bcur = B_OFF + (breg ? B_BYTES : 0u);
    { v8f acc = {};
      gemm_tile(lds_raw, aRow0, bcur, bRow0, XINW, lane, acc);
      store_act(p.e1, Hc, aRow0, n0, lane, acc, p.b_e0, true); }
    breg ^= 1;
    gbar(p.bar, ++bars * NWG);

    // S2: e2 = relu(e1 @ W1 + b1)
    stage_dma(lds_raw, breg, p.e1, Hc, 64, p.W2t + (size_t)nrow * Hc, Hc);
    bcur = B_OFF + (breg ? B_BYTES : 0u);
    { v8f acc = {};
      gemm_tile(lds_raw, aRow0, bcur, bRow0, Hc, lane, acc);
      store_act(p.e2, Hc, aRow0, n0, lane, acc, p.b_e1, true); }
    breg ^= 1;
    gbar(p.bar, ++bars * NWG);

    // S3: e3 = relu(e2 @ W2 + b2)
    stage_dma(lds_raw, breg, p.e2, Hc, 64, p.Wih + (size_t)nrow * Hc, Hc);
    bcur = B_OFF + (breg ? B_BYTES : 0u);
    { v8f acc = {};
      gemm_tile(lds_raw, aRow0, bcur, bRow0, Hc, lane, acc);
      store_act(p.e3, Hc, aRow0, n0, lane, acc, p.b_e2, true); }
    breg ^= 1;
    gbar(p.bar, ++bars * NWG);

    // S4: h = tanh(e3@Wih + h_prev@Whh + bih + bhh); hr = relu(h)
    { v8f acc = {};
      stage_dma(lds_raw, breg, p.e3, Hc, 64, p.Whh + (size_t)nrow * Hc, Hc);
      bcur = B_OFF + (breg ? B_BYTES : 0u);
      gemm_tile(lds_raw, aRow0, bcur, bRow0, Hc, lane, acc);
      breg ^= 1;
      stage_dma(lds_raw, breg, hp, Hc, 64, p.O0t + (size_t)nrow * Hc, Hc);
      bcur = B_OFF + (breg ? B_BYTES : 0u);
      gemm_tile(lds_raw, aRow0, bcur, bRow0, Hc, lane, acc);
      breg ^= 1;
      int n  = n0 + (lane & 15);
      float bv = p.b_ih[n] + p.b_hh[n];
      int rb = aRow0 + ((lane >> 4) << 3);
#pragma unroll
      for (int r = 0; r < 8; ++r) {
        float v = tanhf(acc[r] + bv);
        hn[(size_t)(rb + r) * Hc + n]   = (__bf16)v;
        p.hr[(size_t)(rb + r) * Hc + n] = (__bf16)fmaxf(v, 0.f);
      } }
    gbar(p.bar, ++bars * NWG);

    // S5: o1 = relu(hr @ Ow0 + b0)
    stage_dma(lds_raw, breg, p.hr, Hc, 64, p.O1t + (size_t)nrow * Hc, Hc);
    bcur = B_OFF + (breg ? B_BYTES : 0u);
    { v8f acc = {};
      gemm_tile(lds_raw, aRow0, bcur, bRow0, Hc, lane, acc);
      store_act(p.o1, Hc, aRow0, n0, lane, acc, p.b_o0, true); }
    breg ^= 1;
    gbar(p.bar, ++bars * NWG);

    // S6: o2 = relu(o1 @ Ow1 + b1); prefetch next B: O2 for blocks 0..7
    //     (S7 participants), else W0 for the next timestep.
    { const __bf16* bn; unsigned bnK;
      if (blockIdx.x < 8) { bn = p.O2t + (size_t)nrow * Hc;   bnK = Hc; }
      else                { bn = p.W0t + (size_t)nrow * XINW; bnK = XINW; }
      stage_dma(lds_raw, breg, p.o1, Hc, 64, bn, bnK);
      bcur = B_OFF + (breg ? B_BYTES : 0u);
      v8f acc = {};
      gemm_tile(lds_raw, aRow0, bcur, bRow0, Hc, lane, acc);
      store_act(p.o2, Hc, aRow0, n0, lane, acc, p.b_o1, true);
      breg ^= 1; }
    gbar(p.bar, ++bars * NWG);

    // S7: o = o2 @ Ow2 + b2 (blocks 0..7, same wave mapping, n0 < 256);
    //     other blocks prefill x for t+1.
    if (blockIdx.x < 8) {
      stage_dma(lds_raw, breg, p.o2, Hc, 64, p.W0t + (size_t)nrow * XINW, XINW);
      bcur = B_OFF + (breg ? B_BYTES : 0u);
      v8f acc = {};
      gemm_tile(lds_raw, aRow0, bcur, bRow0, Hc, lane, acc);
      int n  = n0 + (lane & 15);
      float bv = p.b_o2[n];
      int rb = aRow0 + ((lane >> 4) << 3);
#pragma unroll
      for (int r = 0; r < 8; ++r) {
        int b = rb + r;
        float v  = acc[r] + bv;
        float vm = (t < p.xlen[b]) ? v : 0.f;
        p.out[((size_t)b * Tlen + t) * DOUTc + n] = vm;
        __bf16 vb = (__bf16)v;   // carry uses unmasked output
        p.xin[(size_t)b * XINW + DXc + n]         = vb;
        p.xin[(size_t)b * XINW + DXc + DOUTc + n] = vb;
      }
      breg ^= 1;
    } else if (t + 1 < Tlen) {
      int tid2 = (wid - 64) * 32 + lane;
      for (int i = tid2; i < Bsz * DXc; i += (NWAVE - 64) * 32) {
        int b = i >> 8, c = i & 255;
        p.xin[(size_t)b * XINW + c] =
            (__bf16)p.x[((size_t)b * Tlen + (t + 1)) * DXc + c];
      }
    }
    gbar(p.bar, ++bars * NWG);
  }
}

// ---- prep: transpose f32 [K][N] -> bf16 [N][K] ----
__global__ void k_transpose_bf16(const float* __restrict__ W, __bf16* __restrict__ Wt,
                                 int K, int N) {
  int total = K * N;
  for (int i = blockIdx.x * blockDim.x + threadIdx.x; i < total;
       i += gridDim.x * blockDim.x) {
    int k = i / N, n = i - k * N;
    Wt[(size_t)n * K + k] = (__bf16)W[i];
  }
}

__global__ void k_init(unsigned* bar) {
  if (threadIdx.x == 0 && blockIdx.x == 0) *bar = 0u;
}

extern "C" void kernel_launch(void* const* d_in, const int* in_sizes, int n_in,
                              void* d_out, int out_size, void* d_ws, size_t ws_size,
                              hipStream_t stream) {
  Params p;
  p.x    = (const float*)d_in[0];
  p.y0   = (const float*)d_in[1];
  p.y1   = (const float*)d_in[2];
  p.xlen = (const int*)  d_in[3];
  const float* ew0 = (const float*)d_in[4];  p.b_e0 = (const float*)d_in[5];
  const float* ew1 = (const float*)d_in[6];  p.b_e1 = (const float*)d_in[7];
  const float* ew2 = (const float*)d_in[8];  p.b_e2 = (const float*)d_in[9];
  const float* wih = (const float*)d_in[10]; p.b_ih = (const float*)d_in[11];
  const float* whh = (const float*)d_in[12]; p.b_hh = (const float*)d_in[13];
  const float* ow0 = (const float*)d_in[14]; p.b_o0 = (const float*)d_in[15];
  const float* ow1 = (const float*)d_in[16]; p.b_o1 = (const float*)d_in[17];
  const float* ow2 = (const float*)d_in[18]; p.b_o2 = (const float*)d_in[19];
  p.out = (float*)d_out;

  char* ws = (char*)d_ws;
  size_t off = 0;
  auto alloc = [&](size_t bytes) -> void* {
    void* ptr = ws + off;
    off = (off + bytes + 255) & ~(size_t)255;
    return ptr;
  };
  __bf16* W0t = (__bf16*)alloc((size_t)Hc * XINW * 2);
  __bf16* W1t = (__bf16*)alloc((size_t)Hc * Hc * 2);
  __bf16* W2t = (__bf16*)alloc((size_t)Hc * Hc * 2);
  __bf16* Wih = (__bf16*)alloc((size_t)Hc * Hc * 2);
  __bf16* Whh = (__bf16*)alloc((size_t)Hc * Hc * 2);
  __bf16* O0t = (__bf16*)alloc((size_t)Hc * Hc * 2);
  __bf16* O1t = (__bf16*)alloc((size_t)Hc * Hc * 2);
  __bf16* O2t = (__bf16*)alloc((size_t)DOUTc * Hc * 2);
  p.W0t = W0t; p.W1t = W1t; p.W2t = W2t; p.Wih = Wih; p.Whh = Whh;
  p.O0t = O0t; p.O1t = O1t; p.O2t = O2t;

  p.xin = (__bf16*)alloc((size_t)Bsz * XINW * 2);
  p.e1  = (__bf16*)alloc((size_t)Bsz * Hc * 2);
  p.e2  = (__bf16*)alloc((size_t)Bsz * Hc * 2);
  p.e3  = (__bf16*)alloc((size_t)Bsz * Hc * 2);
  p.h0  = (__bf16*)alloc((size_t)Bsz * Hc * 2);
  p.h1  = (__bf16*)alloc((size_t)Bsz * Hc * 2);
  p.hr  = (__bf16*)alloc((size_t)Bsz * Hc * 2);
  p.o1  = (__bf16*)alloc((size_t)Bsz * Hc * 2);
  p.o2  = (__bf16*)alloc((size_t)Bsz * Hc * 2);
  p.bar = (unsigned*)alloc(256);

  dim3 tb(256);
  k_transpose_bf16<<<512, tb, 0, stream>>>(ew0, W0t, XINW, Hc);
  k_transpose_bf16<<<512, tb, 0, stream>>>(ew1, W1t, Hc, Hc);
  k_transpose_bf16<<<512, tb, 0, stream>>>(ew2, W2t, Hc, Hc);
  k_transpose_bf16<<<512, tb, 0, stream>>>(wih, Wih, Hc, Hc);
  k_transpose_bf16<<<512, tb, 0, stream>>>(whh, Whh, Hc, Hc);
  k_transpose_bf16<<<512, tb, 0, stream>>>(ow0, O0t, Hc, Hc);
  k_transpose_bf16<<<512, tb, 0, stream>>>(ow1, O1t, Hc, Hc);
  k_transpose_bf16<<<512, tb, 0, stream>>>(ow2, O2t, Hc, DOUTc);
  k_init<<<1, 64, 0, stream>>>(p.bar);

  rnn_persistent<<<NWG, TPB, LDS_TOTAL, stream>>>(p);
}
